// NemotronFFN_MoE_43946105372963
// MI455X (gfx1250) — matrix operand order repaired and checked by reading, not compiled, not run on hardware
//
#include <hip/hip_runtime.h>
#include <hip/hip_bf16.h>

typedef __attribute__((ext_vector_type(16))) __bf16 v16bf;
typedef __attribute__((ext_vector_type(8)))  float  v8f;
typedef __attribute__((ext_vector_type(16))) unsigned short v16u;
typedef __attribute__((ext_vector_type(4)))  int    v4i;

#define N_TOK   8192
#define DMODEL  2048
#define MFF     1024
#define NEXP    8
#define CAP     16896   /* 2*N_TOK + NEXP*64 : padded assignment capacity */
#define MAXTILE 264     /* CAP / 64 */
#define TM      64
#define TNB     128     /* wide N tile: 8 col-blocks, 4 accumulators/wave */
#define KS      32

// ---------------------------------------------------- CDNA5 async helpers ---
#if defined(__has_builtin)
#if __has_builtin(__builtin_amdgcn_global_load_async_to_lds_b128)
#define HAVE_ASYNC_BUILTIN 1
#endif
#if __has_builtin(__builtin_amdgcn_s_wait_asynccnt)
#define HAVE_WAIT_BUILTIN 1
#endif
#endif

__device__ __forceinline__ void async_copy16(void* lds_dst, const void* gsrc) {
#ifdef HAVE_ASYNC_BUILTIN
  __builtin_amdgcn_global_load_async_to_lds_b128(
      (__attribute__((address_space(1))) v4i*)gsrc,
      (__attribute__((address_space(3))) v4i*)lds_dst, 0, 0);
#else
  asm volatile("global_load_async_to_lds_b128 %0, %1, off"
               :: "v"((unsigned)(unsigned long long)lds_dst),
                  "v"((unsigned long long)gsrc)
               : "memory");
#endif
}

__device__ __forceinline__ void wait_async0() {
#ifdef HAVE_WAIT_BUILTIN
  __builtin_amdgcn_s_wait_asynccnt(0);
#else
  asm volatile("s_wait_asynccnt 0x0" ::: "memory");
#endif
}

// Native bf16 convert (RTE) -- lets the backend use hardware cvt ops.
__device__ __forceinline__ unsigned short f2bf(float f) {
  __bf16 h = (__bf16)f;
  return __builtin_bit_cast(unsigned short, h);
}

// ---------------------------------------------------------------- router ----
__global__ __launch_bounds__(256) void router_kernel(
    const float* __restrict__ x, const float* __restrict__ rw,
    int* __restrict__ topi, float* __restrict__ topw) {
  const int wave = threadIdx.x >> 5;
  const int lane = threadIdx.x & 31;
  const int n = blockIdx.x * 8 + wave;
  if (n >= N_TOK) return;
  const float* xr = x + (size_t)n * DMODEL;
  float logits[NEXP];
  #pragma unroll
  for (int e = 0; e < NEXP; ++e) {
    const float* wr = rw + (size_t)e * DMODEL;
    float p = 0.f;
    for (int k = lane; k < DMODEL; k += 32) p += xr[k] * wr[k];
    #pragma unroll
    for (int off = 16; off >= 1; off >>= 1) p += __shfl_xor(p, off, 32);
    logits[e] = p;
  }
  if (lane == 0) {
    int i1 = 0;
    #pragma unroll
    for (int e = 1; e < NEXP; ++e) if (logits[e] > logits[i1]) i1 = e;
    int i2 = (i1 == 0) ? 1 : 0;
    #pragma unroll
    for (int e = 0; e < NEXP; ++e)
      if (e != i1 && logits[e] > logits[i2]) i2 = e;
    float w2 = 1.f / (1.f + __expf(logits[i1] - logits[i2]));
    float w1 = 1.f - w2;
    topi[2 * n] = i1; topi[2 * n + 1] = i2;
    topw[2 * n] = w1; topw[2 * n + 1] = w2;
  }
}

// ------------------------------------------------------------- utilities ----
__global__ void cvt_bf16_kernel(const float* __restrict__ x,
                                unsigned short* __restrict__ xb, int n) {
  const int stride = gridDim.x * blockDim.x * 4;
  for (int i = (blockIdx.x * blockDim.x + threadIdx.x) * 4; i < n; i += stride) {
    float4 f = *(const float4*)(x + i);
    alignas(8) unsigned short t[4] = {f2bf(f.x), f2bf(f.y), f2bf(f.z), f2bf(f.w)};
    *(uint2*)(xb + i) = *(const uint2*)t;
  }
}

__global__ void zero_counts_kernel(int* cnt) {
  if (threadIdx.x < NEXP) cnt[threadIdx.x] = 0;
}

__global__ void count_kernel(const int* __restrict__ topi, int* cnt) {
  int n = blockIdx.x * blockDim.x + threadIdx.x;
  if (n < N_TOK) {
    atomicAdd(&cnt[topi[2 * n]], 1);
    atomicAdd(&cnt[topi[2 * n + 1]], 1);
  }
}

__global__ void offsets_kernel(const int* __restrict__ cnt, int* cur,
                               int* texp, int* atok, float* agat) {
  __shared__ int soff[NEXP + 1];
  if (threadIdx.x == 0) {
    int acc = 0;
    for (int e = 0; e < NEXP; ++e) {
      soff[e] = acc; cur[e] = acc;
      acc += ((cnt[e] + TM - 1) / TM) * TM;
    }
    soff[NEXP] = acc;
  }
  __syncthreads();
  for (int t = threadIdx.x; t < MAXTILE; t += blockDim.x) {
    int r = t * TM, e = -1;
    for (int q = 0; q < NEXP; ++q)
      if (r >= soff[q] && r < soff[q + 1]) e = q;
    texp[t] = e;
  }
  for (int i = threadIdx.x; i < CAP; i += blockDim.x) {
    atok[i] = -1; agat[i] = 0.f;
  }
}

__global__ void scatter_kernel(const int* __restrict__ topi,
                               const float* __restrict__ topw,
                               int* cur, int* atok, float* agat) {
  int n = blockIdx.x * blockDim.x + threadIdx.x;
  if (n < N_TOK) {
    #pragma unroll
    for (int j = 0; j < 2; ++j) {
      int e = topi[2 * n + j];
      int pos = atomicAdd(&cur[e], 1);
      atok[pos] = n;
      agat[pos] = topw[2 * n + j];
    }
  }
}

// ------------------------------------------- WMMA fragment construction ----
__device__ __forceinline__ v16bf load_afrag(const unsigned short (*T)[KS],
                                            int rbase, int half, int l16) {
  v16u u;
  const unsigned short* row = T[rbase + l16];
  #pragma unroll
  for (int v = 0; v < 8; ++v) {
    const int ka = ((v < 4) ? 2 * v : 16 + 2 * (v - 4)) + half * 8;
    u[2 * v]     = row[ka];
    u[2 * v + 1] = row[ka + 1];
  }
  return __builtin_bit_cast(v16bf, u);
}
__device__ __forceinline__ v16bf load_bfrag(const unsigned short (*T)[KS],
                                            int cbase, int half, int l16) {
  v16u u;
  const unsigned short* col = T[cbase + l16];
  #pragma unroll
  for (int v = 0; v < 8; ++v) {
    const int kb = ((v < 4) ? 2 * v : 16 + 2 * (v - 4)) + half * 8;
    u[2 * v]     = col[kb];
    u[2 * v + 1] = col[kb + 1];
  }
  return __builtin_bit_cast(v16bf, u);
}

// ------------------------------------------------------------ GEMM pass 1 ---
// H[row, :MFF] = relu(X[tok(row), :] @ W1^T)^2  -- async-pipelined bf16 WMMA.
// Padded rows (tok<0) read token 0's data: WMMA contamination is row-local
// and those output rows are discarded by the gate guard in pass 2.
__global__ __launch_bounds__(256) void ffn_gemm1(
    const unsigned short* __restrict__ Xb,   // [N_TOK, DMODEL] bf16
    const float* __restrict__ Wbase,         // [(E,)MFF, DMODEL] f32
    unsigned short* __restrict__ H,          // [rows, MFF] bf16
    const int* __restrict__ atok,
    const int* __restrict__ texp, int moe) {
  __shared__ __align__(16) unsigned short Xs[2][TM][KS];    //  8 KB
  __shared__ __align__(16) unsigned short Ws[2][TNB][KS];   // 16 KB

  const int tile = blockIdx.x;
  const float* W = Wbase;
  if (moe) {
    int e = texp[tile];
    if (e < 0) return;
    W += (size_t)e * (size_t)MFF * (size_t)DMODEL;
  }
  const int row0 = tile * TM;
  const int col0 = blockIdx.y * TNB;

  const int tid = threadIdx.x, lane = tid & 31, wave = tid >> 5;
  const int half = lane >> 4, l16 = lane & 15;
  const int rb = wave & 3, cb0 = (wave >> 2) * 4;
  const int srow = tid >> 2, sseg = (tid & 3) * 8;

  int tok = moe ? atok[row0 + srow] : (row0 + srow);
  if (tok < 0) tok = 0;                       // branchless padded-row handling
  const unsigned short* xrow = Xb + (size_t)tok * DMODEL;
  const float* wrow0 = W + (size_t)(col0 + (tid >> 2)) * DMODEL;
  const float* wrow1 = W + (size_t)(col0 + ((tid + 256) >> 2)) * DMODEL;

  auto stageX = [&](int b, int kk) {
    async_copy16(&Xs[b][srow][sseg], xrow + kk + sseg);
  };
  auto stageW = [&](int b, int kk) {
    const int wr0 = tid >> 2, wr1 = (tid + 256) >> 2, wsg = (tid & 3) * 8;
    alignas(16) unsigned short t[8];
    #pragma unroll
    for (int i = 0; i < 8; ++i) t[i] = f2bf(wrow0[kk + wsg + i]);
    *(uint4*)&Ws[b][wr0][wsg] = *(const uint4*)t;
    #pragma unroll
    for (int i = 0; i < 8; ++i) t[i] = f2bf(wrow1[kk + wsg + i]);
    *(uint4*)&Ws[b][wr1][wsg] = *(const uint4*)t;
  };

  v8f acc[4];
  {
    v8f z = {0, 0, 0, 0, 0, 0, 0, 0};
    #pragma unroll
    for (int c = 0; c < 4; ++c) acc[c] = z;
  }
  auto compute = [&](int b) {
    v16bf a = load_afrag(Xs[b], rb * 16, half, l16);
    #pragma unroll
    for (int c = 0; c < 4; ++c) {
      v16bf bb = load_bfrag(Ws[b], (cb0 + c) * 16, half, l16);
      acc[c] = __builtin_amdgcn_wmma_f32_16x16x32_bf16(
          false, a, false, bb, (short)0, acc[c], false, false);
    }
  };

  // 2-phase software pipeline, static buffer indices, no in-loop guards.
  stageX(0, 0); stageW(0, 0);
  wait_async0(); __syncthreads();
  int kk = 0;
  #pragma unroll 1
  for (int it = 0; it < DMODEL / (2 * KS) - 1; ++it) {
    stageX(1, kk + KS);     stageW(1, kk + KS);
    compute(0);
    wait_async0(); __syncthreads();
    stageX(0, kk + 2 * KS); stageW(0, kk + 2 * KS);
    compute(1);
    wait_async0(); __syncthreads();
    kk += 2 * KS;
  }
  stageX(1, kk + KS); stageW(1, kk + KS);
  compute(0);
  wait_async0(); __syncthreads();
  compute(1);

  #pragma unroll
  for (int r = 0; r < 8; ++r) {
    const int mr = row0 + rb * 16 + half * 8 + r;
    #pragma unroll
    for (int c = 0; c < 4; ++c) {
      float v = acc[c][r]; v = fmaxf(v, 0.f); v *= v;    // relu^2
      H[(size_t)mr * MFF + col0 + (cb0 + c) * 16 + l16] = f2bf(v);
    }
  }
}

// ------------------------------------------------------------ GEMM pass 2 ---
// out[tok(row), :DMODEL] (+)= gate(row) * (H[row, :] @ W2^T)
__global__ __launch_bounds__(256) void ffn_gemm2(
    const unsigned short* __restrict__ H,    // [rows, MFF] bf16
    const float* __restrict__ Wbase,         // [(E,)DMODEL, MFF] f32
    float* __restrict__ out,                 // [N_TOK, DMODEL] f32
    const int* __restrict__ atok,
    const float* __restrict__ agat,
    const int* __restrict__ texp, int moe) {
  __shared__ __align__(16) unsigned short Hs[2][TM][KS];    //  8 KB
  __shared__ __align__(16) unsigned short Ws[2][TNB][KS];   // 16 KB

  const int tile = blockIdx.x;
  const float* W = Wbase;
  if (moe) {
    int e = texp[tile];
    if (e < 0) return;
    W += (size_t)e * (size_t)DMODEL * (size_t)MFF;
  }
  const int row0 = tile * TM;
  const int col0 = blockIdx.y * TNB;

  const int tid = threadIdx.x, lane = tid & 31, wave = tid >> 5;
  const int half = lane >> 4, l16 = lane & 15;
  const int rb = wave & 3, cb0 = (wave >> 2) * 4;
  const int srow = tid >> 2, sseg = (tid & 3) * 8;

  const unsigned short* hrow = H + (size_t)(row0 + srow) * MFF;
  const float* wrow0 = W + (size_t)(col0 + (tid >> 2)) * MFF;
  const float* wrow1 = W + (size_t)(col0 + ((tid + 256) >> 2)) * MFF;

  auto stageA = [&](int b, int kk) {
    async_copy16(&Hs[b][srow][sseg], hrow + kk + sseg);
  };
  auto stageW = [&](int b, int kk) {
    const int wr0 = tid >> 2, wr1 = (tid + 256) >> 2, wsg = (tid & 3) * 8;
    alignas(16) unsigned short t[8];
    #pragma unroll
    for (int i = 0; i < 8; ++i) t[i] = f2bf(wrow0[kk + wsg + i]);
    *(uint4*)&Ws[b][wr0][wsg] = *(const uint4*)t;
    #pragma unroll
    for (int i = 0; i < 8; ++i) t[i] = f2bf(wrow1[kk + wsg + i]);
    *(uint4*)&Ws[b][wr1][wsg] = *(const uint4*)t;
  };

  v8f acc[4];
  {
    v8f z = {0, 0, 0, 0, 0, 0, 0, 0};
    #pragma unroll
    for (int c = 0; c < 4; ++c) acc[c] = z;
  }
  auto compute = [&](int b) {
    v16bf a = load_afrag(Hs[b], rb * 16, half, l16);
    #pragma unroll
    for (int c = 0; c < 4; ++c) {
      v16bf bb = load_bfrag(Ws[b], (cb0 + c) * 16, half, l16);
      acc[c] = __builtin_amdgcn_wmma_f32_16x16x32_bf16(
          false, a, false, bb, (short)0, acc[c], false, false);
    }
  };

  stageA(0, 0); stageW(0, 0);
  wait_async0(); __syncthreads();
  int kk = 0;
  #pragma unroll 1
  for (int it = 0; it < MFF / (2 * KS) - 1; ++it) {
    stageA(1, kk + KS);     stageW(1, kk + KS);
    compute(0);
    wait_async0(); __syncthreads();
    stageA(0, kk + 2 * KS); stageW(0, kk + 2 * KS);
    compute(1);
    wait_async0(); __syncthreads();
    kk += 2 * KS;
  }
  stageA(1, kk + KS); stageW(1, kk + KS);
  compute(0);
  wait_async0(); __syncthreads();
  compute(1);

  #pragma unroll
  for (int r = 0; r < 8; ++r) {
    const int mr = row0 + rb * 16 + half * 8 + r;
    if (moe) {
      int tok = atok[mr];
      if (tok >= 0) {
        float g = agat[mr];
        #pragma unroll
        for (int c = 0; c < 4; ++c) {
          const int cc = col0 + (cb0 + c) * 16 + l16;
          atomicAdd(&out[(size_t)tok * DMODEL + cc], acc[c][r] * g);
        }
      }
    } else {
      #pragma unroll
      for (int c = 0; c < 4; ++c) {
        const int cc = col0 + (cb0 + c) * 16 + l16;
        out[(size_t)mr * DMODEL + cc] = acc[c][r];   // shared expert writes first
      }
    }
  }
}

// ---------------------------------------------------------------- launch ----
extern "C" void kernel_launch(void* const* d_in, const int* in_sizes, int n_in,
                              void* d_out, int out_size, void* d_ws,
                              size_t ws_size, hipStream_t stream) {
  (void)in_sizes; (void)n_in; (void)out_size; (void)ws_size;
  const float* x   = (const float*)d_in[0];
  const float* rw  = (const float*)d_in[1];
  const float* w1s = (const float*)d_in[2];
  const float* w2s = (const float*)d_in[3];
  const float* sw1 = (const float*)d_in[4];
  const float* sw2 = (const float*)d_in[5];
  float* out = (float*)d_out;

  char* ws = (char*)d_ws;
  int*   topi = (int*)(ws + 0);                       // 64 KB
  float* topw = (float*)(ws + (64 << 10));            // 64 KB
  int*   cnt  = (int*)(ws + (128 << 10));             // 8 ints
  int*   cur  = (int*)(ws + (128 << 10) + 64);        // 8 ints
  int*   texp = (int*)(ws + (128 << 10) + 256);       // 264 ints
  int*   atok = (int*)(ws + (132 << 10));             // 66 KB
  float* agat = (float*)(ws + (200 << 10));           // 66 KB
  unsigned short* Xb = (unsigned short*)(ws + (1 << 20));               // 32 MB
  unsigned short* H  = (unsigned short*)(ws + (1 << 20) +
                                         (size_t)N_TOK * DMODEL * 2);   // 33 MB

  router_kernel<<<N_TOK / 8, 256, 0, stream>>>(x, rw, topi, topw);
  cvt_bf16_kernel<<<4096, 256, 0, stream>>>(x, Xb, N_TOK * DMODEL);
  zero_counts_kernel<<<1, 32, 0, stream>>>(cnt);
  count_kernel<<<N_TOK / 256, 256, 0, stream>>>(topi, cnt);
  offsets_kernel<<<1, 256, 0, stream>>>(cnt, cur, texp, atok, agat);
  scatter_kernel<<<N_TOK / 256, 256, 0, stream>>>(topi, topw, cur, atok, agat);

  // shared expert (plain stores) then MoE experts (gate-scaled atomics)
  ffn_gemm1<<<dim3(N_TOK / TM, MFF / TNB), 256, 0, stream>>>(
      Xb, sw1, H, atok, texp, 0);
  ffn_gemm2<<<dim3(N_TOK / TM, DMODEL / TNB), 256, 0, stream>>>(
      H, sw2, out, atok, agat, texp, 0);
  ffn_gemm1<<<dim3(MAXTILE, MFF / TNB), 256, 0, stream>>>(
      Xb, w1s, H, atok, texp, 1);
  ffn_gemm2<<<dim3(MAXTILE, DMODEL / TNB), 256, 0, stream>>>(
      H, w2s, out, atok, agat, texp, 1);
}
